// Qwen3_5VisionAttention_56916906606622
// MI455X (gfx1250) — matrix operand order repaired and checked by reading, not compiled, not run on hardware
//
#include <hip/hip_runtime.h>

typedef __attribute__((ext_vector_type(16))) _Float16 v16h;
typedef __attribute__((ext_vector_type(8)))  _Float16 v8h;
typedef __attribute__((ext_vector_type(8)))  float    v8f;

#define WMMA_F16(A, B, C) \
  __builtin_amdgcn_wmma_f32_16x16x32_f16(false, (A), false, (B), (short)0, (C), false, false)

// ---------------------------------------------------------------------------
// Fragment loaders (wave32, contiguous k order k = 16*(lane>>4) + s, same in A and B)
// A (16x32 f16): lane m = lane&15; kbase = (lane>=16)?16:0;
//   halfs[0..7]  = A[m][kbase .. kbase+7]
//   halfs[8..15] = A[m][kbase+8 .. kbase+15]
// ---------------------------------------------------------------------------
__device__ __forceinline__ v16h load_a16x32(const _Float16* tile, int ld) {
  int lane = threadIdx.x & 31;
  int m  = lane & 15;
  int kb = (lane >> 4) << 4;          // 0 or 16
  const _Float16* p = tile + m * ld + kb;
  v8h lo = *(const v8h*)(p);
  v8h hi = *(const v8h*)(p + 8);
  v16h r;
#pragma unroll
  for (int i = 0; i < 8; ++i) { r[i] = lo[i]; r[i + 8] = hi[i]; }
  return r;
}

// B (32x16 f16) where B[k][n] = T[n][k], T row-major (leading dim ld halfs):
// lane n = lane&15; kbase = (lane>=16)?16:0; halfs = T[n][kbase .. kbase+15]
__device__ __forceinline__ v16h load_bT16x32(const _Float16* t, int ld) {
  int lane = threadIdx.x & 31;
  int n  = lane & 15;
  int kb = (lane >> 4) << 4;          // 0 or 16
  return *(const v16h*)(t + n * ld + kb);
}

// ---------------------------------------------------------------------------
// Elementwise conversion kernels
// ---------------------------------------------------------------------------
__global__ void cvt_f32_to_f16(const float* __restrict__ in, _Float16* __restrict__ out, int n) {
  int i = blockIdx.x * blockDim.x + threadIdx.x;
  if (i < n) out[i] = (_Float16)in[i];
}

// in: [rows][cols] f32  ->  out: [cols][rows] f16  (coalesced writes)
__global__ void transpose_f32_to_f16(const float* __restrict__ in, _Float16* __restrict__ out,
                                     int rows, int cols) {
  int i = blockIdx.x * blockDim.x + threadIdx.x;
  if (i >= rows * cols) return;
  int r = i % rows;
  int c = i / rows;
  out[(size_t)c * rows + r] = (_Float16)in[(size_t)r * cols + c];
}

// ---------------------------------------------------------------------------
// WMMA GEMM:  C[M][N] = A[M][K] * BT[N][K]^T + bias
// block = 128 threads (4 waves, 2x2), block tile 128x128, wave tile 64x64
// (16 accumulators, 8 fragment loads per k-step -> 32 FLOP/byte from cache)
// ---------------------------------------------------------------------------
__global__ void __launch_bounds__(128)
wmma_gemm_bias(const _Float16* __restrict__ A, const _Float16* __restrict__ BT,
               const float* __restrict__ bias, _Float16* __restrict__ outh,
               float* __restrict__ outf, int M, int N, int K) {
  int wid  = threadIdx.x >> 5;
  int lane = threadIdx.x & 31;
  int m0 = blockIdx.y * 128 + (wid & 1) * 64;
  int n0 = blockIdx.x * 128 + (wid >> 1) * 64;

  v8f acc[4][4] = {};
  for (int k0 = 0; k0 < K; k0 += 32) {
    v16h a[4], b[4];
#pragma unroll
    for (int i = 0; i < 4; ++i)
      a[i] = load_a16x32(A + (size_t)(m0 + 16 * i) * K + k0, K);
#pragma unroll
    for (int i = 0; i < 4; ++i)
      b[i] = load_bT16x32(BT + (size_t)(n0 + 16 * i) * K + k0, K);
#pragma unroll
    for (int mt = 0; mt < 4; ++mt)
#pragma unroll
      for (int nt = 0; nt < 4; ++nt)
        acc[mt][nt] = WMMA_F16(a[mt], b[nt], acc[mt][nt]);
  }

  int hi = lane >> 4, nn = lane & 15;
#pragma unroll
  for (int mt = 0; mt < 4; ++mt)
#pragma unroll
    for (int nt = 0; nt < 4; ++nt) {
      int col = n0 + nt * 16 + nn;
      float bv = bias[col];
#pragma unroll
      for (int r = 0; r < 8; ++r) {
        int row = m0 + mt * 16 + r + 8 * hi;
        float v = acc[mt][nt][r] + bv;
        if (outh) outh[(size_t)row * N + col] = (_Float16)v;
        else      outf[(size_t)row * N + col] = v;
      }
    }
}

// ---------------------------------------------------------------------------
// RoPE + scatter:  qkvh[S][3840] -> qh/kh [c][h][1024][96] (D padded 80->96),
//                                  vT    [c][h][80][1024]
// ---------------------------------------------------------------------------
__global__ void rope_scatter(const _Float16* __restrict__ qkvh,
                             const float* __restrict__ cosp, const float* __restrict__ sinp,
                             _Float16* __restrict__ qh, _Float16* __restrict__ kh,
                             _Float16* __restrict__ vT) {
  int idx = blockIdx.x * blockDim.x + threadIdx.x;
  if (idx >= 8192 * 16 * 96) return;
  int dp = idx % 96;
  int h  = (idx / 96) & 15;
  int s  = idx / (96 * 16);
  int c = s >> 10, l = s & 1023;
  size_t od = ((size_t)(c * 16 + h) * 1024 + l) * 96 + dp;
  float qo = 0.f, ko = 0.f;
  if (dp < 80) {
    int d = dp;
    size_t rb = (size_t)s * 3840 + h * 80;
    float q = (float)qkvh[rb + d];
    float k = (float)qkvh[rb + 1280 + d];
    int   d2  = (d < 40) ? d + 40 : d - 40;
    float sgn = (d < 40) ? -1.f : 1.f;
    float qr = sgn * (float)qkvh[rb + d2];
    float kr = sgn * (float)qkvh[rb + 1280 + d2];
    float cv = cosp[s * 80 + d], sv = sinp[s * 80 + d];
    qo = q * cv + qr * sv;
    ko = k * cv + kr * sv;
    vT[((size_t)(c * 16 + h) * 80 + d) * 1024 + l] = qkvh[rb + 2560 + d];
  }
  qh[od] = (_Float16)qo;
  kh[od] = (_Float16)ko;
}

// ---------------------------------------------------------------------------
// Flash attention: one block per (c, h, 64-query tile); 4 waves x 16 rows.
// Double-buffered K/V staging in LDS with register prefetch: next tile's
// global loads are issued before the current tile's WMMAs, ds_store'd into
// the other buffer, one barrier per iteration.
// ---------------------------------------------------------------------------
__global__ void __launch_bounds__(128)
attn_kernel(const _Float16* __restrict__ qh, const _Float16* __restrict__ kh,
            const _Float16* __restrict__ vT, _Float16* __restrict__ attnh) {
  int qb = blockIdx.x & 15;
  int h  = (blockIdx.x >> 4) & 15;
  int c  = blockIdx.x >> 8;
  int wid  = threadIdx.x >> 5;
  int lane = threadIdx.x & 31;
  int tid  = threadIdx.x;

  __shared__ _Float16 ktile[2][32 * 96] __attribute__((aligned(64)));
  __shared__ _Float16 vtile[2][80 * 32] __attribute__((aligned(64)));
  __shared__ _Float16 ptile[4][16 * 32] __attribute__((aligned(64)));

  const size_t chq = (size_t)(c * 16 + h) * 1024 * 96;
  const _Float16* qbase = qh + chq + (size_t)(qb * 64 + wid * 16) * 96;
  const _Float16* kbase = kh + chq;
  const _Float16* vbase = vT + (size_t)(c * 16 + h) * 80 * 1024;

  // K tile: 32x96 halfs = 384 v8h chunks, 3 per thread.
  // V tile: 80x32 halfs = 320 v8h chunks, 2 per thread + 1 for tid<64.
  v8h kreg[3], vreg[3];
  auto load_regs = [&](int j) {
#pragma unroll
    for (int t = 0; t < 3; ++t) {
      int i = tid + t * 128, row = i / 12, part = i % 12;
      kreg[t] = *(const v8h*)(kbase + (size_t)(j + row) * 96 + part * 8);
    }
#pragma unroll
    for (int t = 0; t < 2; ++t) {
      int i = tid + t * 128, row = i >> 2, part = i & 3;
      vreg[t] = *(const v8h*)(vbase + (size_t)row * 1024 + j + part * 8);
    }
    if (tid < 64) {
      int i = tid + 256, row = i >> 2, part = i & 3;
      vreg[2] = *(const v8h*)(vbase + (size_t)row * 1024 + j + part * 8);
    }
  };
  auto store_regs = [&](int buf) {
    _Float16* kt = ktile[buf];
    _Float16* vt = vtile[buf];
#pragma unroll
    for (int t = 0; t < 3; ++t) {
      int i = tid + t * 128, row = i / 12, part = i % 12;
      *(v8h*)(kt + row * 96 + part * 8) = kreg[t];
    }
#pragma unroll
    for (int t = 0; t < 2; ++t) {
      int i = tid + t * 128, row = i >> 2, part = i & 3;
      *(v8h*)(vt + row * 32 + part * 8) = vreg[t];
    }
    if (tid < 64) {
      int i = tid + 256, row = i >> 2, part = i & 3;
      *(v8h*)(vt + row * 32 + part * 8) = vreg[2];
    }
  };

  v16h aq[3];
#pragma unroll
  for (int kc = 0; kc < 3; ++kc) aq[kc] = load_a16x32(qbase + kc * 32, 96);

  v8f oacc[5] = {};
  float mrow[8], lrow[8];
#pragma unroll
  for (int r = 0; r < 8; ++r) { mrow[r] = -1e30f; lrow[r] = 0.f; }
  const float c1 = 1.4426950408889634f / 8.94427190999916f;  // log2(e)/sqrt(80)
  int hi = lane >> 4, nn = lane & 15;

  // prologue: stage tile 0
  load_regs(0);
  store_regs(0);
  __syncthreads();

  for (int j0 = 0; j0 < 1024; j0 += 32) {
    int cur = (j0 >> 5) & 1;
    bool have_next = (j0 + 32) < 1024;
    if (have_next) load_regs(j0 + 32);  // issue next tile's global loads early

    // S = Q * K^T  (two 16x16 tiles covering kv cols j0..j0+31)
    const _Float16* kt = ktile[cur];
    const _Float16* vt = vtile[cur];
    v8f s0 = {}, s1 = {};
#pragma unroll
    for (int kc = 0; kc < 3; ++kc) {
      v16h b0 = load_bT16x32(kt + kc * 32, 96);
      v16h b1 = load_bT16x32(kt + 16 * 96 + kc * 32, 96);
      s0 = WMMA_F16(aq[kc], b0, s0);
      s1 = WMMA_F16(aq[kc], b1, s1);
    }

    // online softmax update (row r+8*hi lives in this lane's half-wave)
    float p0[8], p1[8];
#pragma unroll
    for (int r = 0; r < 8; ++r) {
      float mj = fmaxf(s0[r], s1[r]);
      mj = fmaxf(mj, __shfl_xor(mj, 1));
      mj = fmaxf(mj, __shfl_xor(mj, 2));
      mj = fmaxf(mj, __shfl_xor(mj, 4));
      mj = fmaxf(mj, __shfl_xor(mj, 8));
      float mn = fmaxf(mrow[r], mj);
      float alpha = exp2f((mrow[r] - mn) * c1);
      p0[r] = exp2f((s0[r] - mn) * c1);
      p1[r] = exp2f((s1[r] - mn) * c1);
      float ps = p0[r] + p1[r];
      ps += __shfl_xor(ps, 1);
      ps += __shfl_xor(ps, 2);
      ps += __shfl_xor(ps, 4);
      ps += __shfl_xor(ps, 8);
      lrow[r] = lrow[r] * alpha + ps;
      mrow[r] = mn;
#pragma unroll
      for (int t = 0; t < 5; ++t) oacc[t][r] *= alpha;
    }

    // stage probs (16x32 f16, row-major, wave-private; DScnt orders WAR/RAW)
    _Float16* pw = ptile[wid];
#pragma unroll
    for (int r = 0; r < 8; ++r) {
      int m = r + 8 * hi;
      pw[m * 32 + nn]      = (_Float16)p0[r];
      pw[m * 32 + 16 + nn] = (_Float16)p1[r];
    }

    v16h pa = load_a16x32(ptile[wid], 32);
#pragma unroll
    for (int t = 0; t < 5; ++t) {
      v16h bv = load_bT16x32(vt + t * 16 * 32, 32);
      oacc[t] = WMMA_F16(pa, bv, oacc[t]);
    }

    // stage next tile into the other buffer; single barrier per iteration
    if (have_next) store_regs(1 - cur);
    __syncthreads();
  }

  // epilogue: attnh[s][h*80 + d] = oacc / l
  int srow0 = c * 1024 + qb * 64 + wid * 16;
#pragma unroll
  for (int t = 0; t < 5; ++t) {
    int col = h * 80 + t * 16 + nn;
#pragma unroll
    for (int r = 0; r < 8; ++r) {
      int row = srow0 + r + 8 * hi;
      attnh[(size_t)row * 1280 + col] = (_Float16)(oacc[t][r] / lrow[r]);
    }
  }
}

// ---------------------------------------------------------------------------
// Host launcher
// ---------------------------------------------------------------------------
extern "C" void kernel_launch(void* const* d_in, const int* in_sizes, int n_in,
                              void* d_out, int out_size, void* d_ws, size_t ws_size,
                              hipStream_t stream) {
  (void)in_sizes; (void)n_in; (void)out_size; (void)ws_size;
  const float* x     = (const float*)d_in[0];
  const float* cosp  = (const float*)d_in[1];
  const float* sinp  = (const float*)d_in[2];
  const float* Wqkv  = (const float*)d_in[3];
  const float* bqkv  = (const float*)d_in[4];
  const float* Wproj = (const float*)d_in[5];
  const float* bproj = (const float*)d_in[6];
  float* out = (float*)d_out;

  const int S = 8192, HID = 1280, N3 = 3840;

  char* ws = (char*)d_ws;
  size_t off = 0;
  auto alloc = [&](size_t bytes) -> char* {
    char* p = ws + off;
    off += (bytes + 255) & ~(size_t)255;
    return p;
  };
  _Float16* xh     = (_Float16*)alloc((size_t)S * HID * 2);            // 21.0 MB
  _Float16* wqkvT  = (_Float16*)alloc((size_t)N3 * HID * 2);           //  9.8 MB
  _Float16* wprojT = (_Float16*)alloc((size_t)HID * HID * 2);          //  3.3 MB
  _Float16* qkvh   = (_Float16*)alloc((size_t)S * N3 * 2);             // 62.9 MB
  _Float16* qhb    = (_Float16*)alloc((size_t)8 * 16 * 1024 * 96 * 2); // 25.2 MB
  _Float16* khb    = (_Float16*)alloc((size_t)8 * 16 * 1024 * 96 * 2); // 25.2 MB
  _Float16* vTb    = (_Float16*)alloc((size_t)8 * 16 * 80 * 1024 * 2); // 21.0 MB
  _Float16* attnh  = (_Float16*)alloc((size_t)S * HID * 2);            // 21.0 MB

  // 1) precision conversion / weight transposes
  {
    int n = S * HID;
    cvt_f32_to_f16<<<(n + 255) / 256, 256, 0, stream>>>(x, xh, n);
  }
  {
    int n = HID * N3;
    transpose_f32_to_f16<<<(n + 255) / 256, 256, 0, stream>>>(Wqkv, wqkvT, HID, N3);
  }
  {
    int n = HID * HID;
    transpose_f32_to_f16<<<(n + 255) / 256, 256, 0, stream>>>(Wproj, wprojT, HID, HID);
  }

  // 2) qkv = x @ Wqkv + bqkv   (f16 out)
  wmma_gemm_bias<<<dim3(N3 / 128, S / 128), 128, 0, stream>>>(
      xh, wqkvT, bqkv, qkvh, nullptr, S, N3, HID);

  // 3) rotary + layout scatter
  {
    int n = S * 16 * 96;
    rope_scatter<<<(n + 255) / 256, 256, 0, stream>>>(qkvh, cosp, sinp, qhb, khb, vTb);
  }

  // 4) chunked attention (8 chunks x 16 heads x 16 query tiles)
  attn_kernel<<<2048, 128, 0, stream>>>(qhb, khb, vTb, attnh);

  // 5) out = attn @ Wproj + bproj   (f32 out)
  wmma_gemm_bias<<<dim3(HID / 128, S / 128), 128, 0, stream>>>(
      attnh, wprojT, bproj, nullptr, out, S, HID, HID);
}
